// RoPEGQAAttn_70033736728864
// MI455X (gfx1250) — matrix operand order repaired and checked by reading, not compiled, not run on hardware
//
#include <hip/hip_runtime.h>
#include <hip/hip_bf16.h>
#include <math.h>

typedef __attribute__((ext_vector_type(16))) __bf16 bf16x16;
typedef __attribute__((ext_vector_type(8)))  __bf16 bf16x8;
typedef __attribute__((ext_vector_type(4)))  __bf16 bf16x4;
typedef __attribute__((ext_vector_type(8)))  float  f32x8;

#define DMODEL 2880
#define NHEADS 64
#define NKV    8
#define DHEAD  64
#define WINDOW 128
#define TSEQ   1536
#define QDIM   4096              // NHEADS*DHEAD
#define KVDIM  512               // NKV*DHEAD
#define QKVD   5120              // QDIM + 2*KVDIM

__device__ __forceinline__ __bf16 f2bf(float f) {
  union { float f; unsigned u; } a; a.f = f;
  unsigned r = a.u + 0x7FFFu + ((a.u >> 16) & 1u);   // round-to-nearest-even
  union { unsigned short s; __bf16 b; } c; c.s = (unsigned short)(r >> 16);
  return c.b;
}

__device__ __forceinline__ bf16x16 join8(bf16x8 lo, bf16x8 hi) {
  return __builtin_shufflevector(lo, hi, 0, 1, 2, 3, 4, 5, 6, 7,
                                         8, 9, 10, 11, 12, 13, 14, 15);
}

__device__ __forceinline__ bf16x8 bf8zero() {
  bf16x8 z;
#pragma unroll
  for (int j = 0; j < 8; ++j) z[j] = f2bf(0.0f);
  return z;
}

// LDS byte address of a __shared__ object: generic shared pointers are
// aperture|lds_offset, so the low 32 bits are the LDS address (ISA 10.2).
__device__ __forceinline__ unsigned lds_addr(const void* p) {
  return (unsigned)(unsigned long long)p;
}

// Async 16B copy global -> LDS (ASYNCcnt-tracked, no VGPR round trip).
__device__ __forceinline__ void async_copy_b128(unsigned lds, unsigned gvoff,
                                                unsigned long long gbase) {
  asm volatile("global_load_async_to_lds_b128 %0, %1, %2"
               :: "v"(lds), "v"(gvoff), "s"(gbase) : "memory");
}
__device__ __forceinline__ void wait_asynccnt0() {
  asm volatile("s_wait_asynccnt 0x0" ::: "memory");
}

// ---------------------------------------------------------------- cvt ------
__global__ __launch_bounds__(256)
void cvt_f32_to_bf16(const float* __restrict__ in, __bf16* __restrict__ out, int n8) {
  int i = blockIdx.x * 256 + threadIdx.x;
  if (i < n8) {
    const float4* p = (const float4*)in + 2 * (size_t)i;
    float4 lo = p[0], hi = p[1];
    bf16x8 v;
    v[0] = f2bf(lo.x); v[1] = f2bf(lo.y); v[2] = f2bf(lo.z); v[3] = f2bf(lo.w);
    v[4] = f2bf(hi.x); v[5] = f2bf(hi.y); v[6] = f2bf(hi.z); v[7] = f2bf(hi.w);
    *((bf16x8*)out + (size_t)i) = v;
  }
}

// ------------------------------------------------------- bf16 WMMA GEMM ----
// C[M,N](f32) = A[M,K]*B[K,N] + bias. Block tile 128x64, KSTEP 64,
// 128 threads = 4 waves; wave w owns rows 32w..32w+31 (16 WMMAs/stage).
// Double-buffered LDS: next tile staged (A via async-to-LDS b128, B
// transposed via packed b64 stores) while current tile is consumed by the
// matrix pipe; one s_wait_asynccnt + one barrier per K-step.
__global__ __launch_bounds__(128)
void gemm_bf16_wmma(const __bf16* __restrict__ A, const __bf16* __restrict__ B,
                    const float* __restrict__ bias, float* __restrict__ C,
                    int M, int N, int K) {
  __shared__ __bf16 As[2][128][72];   // 128x64 tiles, 144B stride
  __shared__ __bf16 BsT[2][64][72];   // transposed 64x64 tiles: BsT[n][k]

  const int tid  = threadIdx.x;
  const int lane = tid & 31;
  const int wid  = tid >> 5;
  const int hf   = lane >> 4;
  const int lm   = lane & 15;
  const int m0   = blockIdx.y * 128;
  const int n0   = blockIdx.x * 64;
  const unsigned long long Abase = (unsigned long long)A;

  auto stage = [&](int buf, int kk) {
    // A tile: 128 rows x 8 chunks of 8 bf16, async global->LDS
#pragma unroll
    for (int it = 0; it < 8; ++it) {
      int idx = tid + 128 * it;
      int r = idx >> 3, cb = (idx & 7) * 8;
      unsigned lds = lds_addr(&As[buf][r][cb]);
      unsigned gvo = (unsigned)(((unsigned)(m0 + r) * (unsigned)K +
                                 (unsigned)(kk + cb)) * 2u);
      async_copy_b128(lds, gvo, Abase);
    }
    // B tile transposed: each thread 4 rows x 8 cols, b64 LDS stores
    int k4 = (tid >> 3) * 4;          // 0..60
    int cb = (tid & 7) * 8;
    const __bf16* bp = B + (size_t)(kk + k4) * N + (n0 + cb);
    bf16x8 r0 = *(const bf16x8*)(bp);
    bf16x8 r1 = *(const bf16x8*)(bp + (size_t)N);
    bf16x8 r2 = *(const bf16x8*)(bp + 2 * (size_t)N);
    bf16x8 r3 = *(const bf16x8*)(bp + 3 * (size_t)N);
#pragma unroll
    for (int j = 0; j < 8; ++j) {
      bf16x4 w; w[0] = r0[j]; w[1] = r1[j]; w[2] = r2[j]; w[3] = r3[j];
      *(bf16x4*)&BsT[buf][cb + j][k4] = w;
    }
  };

  f32x8 acc[2][4];
#pragma unroll
  for (int mt = 0; mt < 2; ++mt)
#pragma unroll
    for (int nt = 0; nt < 4; ++nt) acc[mt][nt] = (f32x8){0,0,0,0,0,0,0,0};

  stage(0, 0);
  wait_asynccnt0();
  __syncthreads();

  const int nsteps = K >> 6;
  for (int s = 0; s < nsteps; ++s) {
    const int cur = s & 1;
    if (s + 1 < nsteps) stage(cur ^ 1, (s + 1) * 64);

    const __bf16 (*Ac)[72]  = As[cur];
    const __bf16 (*BcT)[72] = BsT[cur];
#pragma unroll
    for (int mt = 0; mt < 2; ++mt) {
      int row = 32 * wid + 16 * mt + lm;
#pragma unroll
      for (int kc = 0; kc < 2; ++kc) {
        bf16x16 a = join8(*(const bf16x8*)&Ac[row][32 * kc + 8 * hf],
                          *(const bf16x8*)&Ac[row][32 * kc + 16 + 8 * hf]);
#pragma unroll
        for (int nt = 0; nt < 4; ++nt) {
          const __bf16* bp2 = &BcT[16 * nt + lm][32 * kc + 8 * hf];
          bf16x16 b = join8(*(const bf16x8*)bp2, *(const bf16x8*)(bp2 + 16));
          acc[mt][nt] = __builtin_amdgcn_wmma_f32_16x16x32_bf16(
              false, a, false, b, (short)0, acc[mt][nt], false, false);
        }
      }
    }
    wait_asynccnt0();     // next tile's async LDS writes have landed
    __syncthreads();      // all waves done reading cur / writing nxt
  }

#pragma unroll
  for (int mt = 0; mt < 2; ++mt)
#pragma unroll
    for (int nt = 0; nt < 4; ++nt) {
      int col = n0 + 16 * nt + lm;
      float bv = bias ? bias[col] : 0.0f;
#pragma unroll
      for (int r = 0; r < 8; ++r) {
        int row = m0 + 32 * wid + 16 * mt + r + 8 * hf;
        C[(size_t)row * N + col] = acc[mt][nt][r] + bv;
      }
    }
}

// ------------------------------------------------- RoPE + split + cast -----
__global__ __launch_bounds__(256)
void rope_split(const float* __restrict__ qkv, const float* __restrict__ bqkv,
                __bf16* __restrict__ qb, __bf16* __restrict__ kb,
                __bf16* __restrict__ vb) {
  const int t   = blockIdx.x;
  const int tid = threadIdx.x;
  const float pos = (float)t / 32.0f;                    // ROPE_SCALE
  const float l2b = log2f(150000.0f);                    // ROPE_BASE
  const float* row = qkv + (size_t)t * QKVD;

  for (int p = tid; p < NHEADS * (DHEAD / 2); p += 256) {
    int h = p >> 5, i = p & 31;
    int off = h * DHEAD + 2 * i;
    float x1 = row[off]     + bqkv[off];
    float x2 = row[off + 1] + bqkv[off + 1];
    float invf = exp2f(-((float)(2 * i) / (float)DHEAD) * l2b);
    float fr = pos * invf;
    float sn = __sinf(fr), cs = __cosf(fr);
    size_t dst = ((size_t)h * TSEQ + t) * DHEAD + 2 * i;
    qb[dst]     = f2bf(x1 * cs - x2 * sn);
    qb[dst + 1] = f2bf(x1 * sn + x2 * cs);
  }
  for (int p = tid; p < NKV * (DHEAD / 2); p += 256) {
    int h = p >> 5, i = p & 31;
    int off = QDIM + h * DHEAD + 2 * i;
    float x1 = row[off]     + bqkv[off];
    float x2 = row[off + 1] + bqkv[off + 1];
    float invf = exp2f(-((float)(2 * i) / (float)DHEAD) * l2b);
    float fr = pos * invf;
    float sn = __sinf(fr), cs = __cosf(fr);
    size_t dst = ((size_t)h * TSEQ + t) * DHEAD + 2 * i;
    kb[dst]     = f2bf(x1 * cs - x2 * sn);
    kb[dst + 1] = f2bf(x1 * sn + x2 * cs);
  }
  for (int e = tid; e < KVDIM; e += 256) {
    int h = e >> 6, d = e & 63;
    int off = QDIM + KVDIM + e;
    vb[((size_t)h * TSEQ + t) * DHEAD + d] = f2bf(row[off] + bqkv[off]);
  }
}

// ----------------------------------------- windowed attention with sink ----
__global__ __launch_bounds__(32)
void attn_wmma(const __bf16* __restrict__ qb, const __bf16* __restrict__ kb,
               const __bf16* __restrict__ vb, const float* __restrict__ sink,
               __bf16* __restrict__ yb) {
  __shared__ __bf16 Qs[16][72];
  __shared__ __bf16 Ks[16][72];
  __shared__ float  Ls[16][160];
  __shared__ __bf16 Ps[16][168];
  __shared__ __bf16 VsT[64][168];

  const int qt   = blockIdx.x;
  const int h    = blockIdx.y;
  const int kvh  = h >> 3;          // rep = 8
  const int t0   = qt * 16;
  const int lane = threadIdx.x;
  const int hf   = lane >> 4;
  const int lm   = lane & 15;
  const int base = t0 - WINDOW;     // key index of column 0

#pragma unroll
  for (int it = 0; it < 4; ++it) {
    int idx = lane + 32 * it;
    int r = idx >> 3, cb = (idx & 7) * 8;
    *(bf16x8*)&Qs[r][cb] =
        *(const bf16x8*)(qb + ((size_t)h * TSEQ + (t0 + r)) * DHEAD + cb);
  }
  // V window transposed: 40 groups of 4 keys x 8 d-chunks; b64 stores
#pragma unroll
  for (int it = 0; it < 10; ++it) {
    int idx = lane + 32 * it;
    int c4 = (idx >> 3) * 4;
    int db = (idx & 7) * 8;
    bf16x8 rr[4];
#pragma unroll
    for (int jj = 0; jj < 4; ++jj) {
      int j = base + c4 + jj;
      rr[jj] = (j >= 0 && j < TSEQ)
               ? *(const bf16x8*)(vb + ((size_t)kvh * TSEQ + j) * DHEAD + db)
               : bf8zero();
    }
#pragma unroll
    for (int d = 0; d < 8; ++d) {
      bf16x4 w; w[0] = rr[0][d]; w[1] = rr[1][d]; w[2] = rr[2][d]; w[3] = rr[3][d];
      *(bf16x4*)&VsT[db + d][c4] = w;
    }
  }
  __syncthreads();

  bf16x16 aq[2];
#pragma unroll
  for (int kc = 0; kc < 2; ++kc) {
    bf16x8 a0 = *(const bf16x8*)&Qs[lm][32 * kc + 8 * hf];
    bf16x8 a1 = *(const bf16x8*)&Qs[lm][32 * kc + 16 + 8 * hf];
    aq[kc] = join8(a0, a1);
  }

  // ---- S = Q K^T over 9 key tiles ----
  for (int tt = 0; tt < 9; ++tt) {
    int kt0 = base + 16 * tt;
#pragma unroll
    for (int it = 0; it < 4; ++it) {
      int idx = lane + 32 * it;
      int r = idx >> 3, cb = (idx & 7) * 8;
      int j = kt0 + r;
      bf16x8 v = (j >= 0 && j < TSEQ)
                 ? *(const bf16x8*)(kb + ((size_t)kvh * TSEQ + j) * DHEAD + cb)
                 : bf8zero();
      *(bf16x8*)&Ks[r][cb] = v;
    }
    __syncthreads();

    f32x8 s8 = (f32x8){0,0,0,0,0,0,0,0};
#pragma unroll
    for (int kc = 0; kc < 2; ++kc) {
      const __bf16* bp = &Ks[lm][32 * kc + 8 * hf];
      bf16x16 b = join8(*(const bf16x8*)bp, *(const bf16x8*)(bp + 16));
      s8 = __builtin_amdgcn_wmma_f32_16x16x32_bf16(
          false, aq[kc], false, b, (short)0, s8, false, false);
    }

#pragma unroll
    for (int r = 0; r < 8; ++r) {
      int m = r + 8 * hf;
      int i = t0 + m;
      int j = kt0 + lm;
      float lg = s8[r] * 0.125f;                       // 1/sqrt(64)
      if (j < 0 || j > i || (i - j) > WINDOW) lg = -1e30f;
      Ls[m][16 * tt + lm] = lg;
    }
    __syncthreads();
  }

  // ---- softmax with sink in denominator (logaddexp semantics) ----
  float sk = sink[h];
  if (lane < 16) {
    int m = lane;
    float mx = -1e30f;
    for (int c = 0; c < 144; ++c) mx = fmaxf(mx, Ls[m][c]);
    mx = fmaxf(mx, sk);
    float sum = 0.0f;
    for (int c = 0; c < 144; ++c) sum += __expf(Ls[m][c] - mx);
    float inv = 1.0f / (sum + __expf(sk - mx));
    for (int c = 0; c < 144; ++c) Ps[m][c] = f2bf(__expf(Ls[m][c] - mx) * inv);
    for (int c = 144; c < 160; ++c) Ps[m][c] = f2bf(0.0f);
  }
  __syncthreads();

  // ---- O = P V ----
#pragma unroll
  for (int nt = 0; nt < 4; ++nt) {
    f32x8 o = (f32x8){0,0,0,0,0,0,0,0};
#pragma unroll
    for (int ch = 0; ch < 5; ++ch) {
      bf16x16 a = join8(*(const bf16x8*)&Ps[lm][32 * ch + 8 * hf],
                        *(const bf16x8*)&Ps[lm][32 * ch + 16 + 8 * hf]);
      const __bf16* bp = &VsT[16 * nt + lm][32 * ch + 8 * hf];
      bf16x16 b = join8(*(const bf16x8*)bp, *(const bf16x8*)(bp + 16));
      o = __builtin_amdgcn_wmma_f32_16x16x32_bf16(
          false, a, false, b, (short)0, o, false, false);
    }
#pragma unroll
    for (int r = 0; r < 8; ++r) {
      int m = r + 8 * hf;
      yb[((size_t)(t0 + m)) * QDIM + h * DHEAD + 16 * nt + lm] = f2bf(o[r]);
    }
  }
}

// ---------------------------------------------------------------- launch ---
extern "C" void kernel_launch(void* const* d_in, const int* in_sizes, int n_in,
                              void* d_out, int out_size, void* d_ws, size_t ws_size,
                              hipStream_t stream) {
  const float* x    = (const float*)d_in[0];
  const float* Wqkv = (const float*)d_in[1];
  const float* bqkv = (const float*)d_in[2];
  const float* Wout = (const float*)d_in[3];
  const float* bout = (const float*)d_in[4];
  const float* sink = (const float*)d_in[5];
  float* out = (float*)d_out;

  char* ws = (char*)d_ws;
  size_t off = 0;
  auto carve = [&](size_t bytes) {
    void* p = ws + off;
    off = (off + bytes + 255) & ~(size_t)255;
    return p;
  };
  __bf16* xb    = (__bf16*)carve((size_t)TSEQ * DMODEL * 2);
  __bf16* Wqkvb = (__bf16*)carve((size_t)DMODEL * QKVD * 2);
  __bf16* Woutb = (__bf16*)carve((size_t)QDIM * DMODEL * 2);
  float*  qkvf  = (float*) carve((size_t)TSEQ * QKVD * 4);
  __bf16* qbuf  = (__bf16*)carve((size_t)NHEADS * TSEQ * DHEAD * 2);
  __bf16* kbuf  = (__bf16*)carve((size_t)NKV * TSEQ * DHEAD * 2);
  __bf16* vbuf  = (__bf16*)carve((size_t)NKV * TSEQ * DHEAD * 2);
  __bf16* ybuf  = (__bf16*)carve((size_t)TSEQ * QDIM * 2);

  int n8;
  n8 = TSEQ * DMODEL / 8;
  cvt_f32_to_bf16<<<(n8 + 255) / 256, 256, 0, stream>>>(x, xb, n8);
  n8 = DMODEL * QKVD / 8;
  cvt_f32_to_bf16<<<(n8 + 255) / 256, 256, 0, stream>>>(Wqkv, Wqkvb, n8);
  n8 = QDIM * DMODEL / 8;
  cvt_f32_to_bf16<<<(n8 + 255) / 256, 256, 0, stream>>>(Wout, Woutb, n8);

  gemm_bf16_wmma<<<dim3(QKVD / 64, TSEQ / 128), 128, 0, stream>>>(
      xb, Wqkvb, nullptr, qkvf, TSEQ, QKVD, DMODEL);

  rope_split<<<TSEQ, 256, 0, stream>>>(qkvf, bqkv, qbuf, kbuf, vbuf);

  attn_wmma<<<dim3(TSEQ / 16, NHEADS), 32, 0, stream>>>(
      qbuf, kbuf, vbuf, sink, ybuf);

  gemm_bf16_wmma<<<dim3(DMODEL / 64, TSEQ / 128), 128, 0, stream>>>(
      ybuf, Woutb, bout, out, TSEQ, DMODEL, QDIM);
}
